// Pointnet2MSG_19404662244024
// MI455X (gfx1250) — matrix operand order repaired and checked by reading, not compiled, not run on hardware
//
#include <hip/hip_runtime.h>
#include <hip/hip_bf16.h>
#include <math.h>

// ---------------------------------------------------------------------------
// PointNet++ MSG forward for MI455X (gfx1250), wave32 + v_wmma_f32_16x16x32_bf16
// ---------------------------------------------------------------------------

typedef __attribute__((ext_vector_type(16))) __bf16        v16bf;
typedef __attribute__((ext_vector_type(8)))  float         v8f;
typedef __attribute__((ext_vector_type(4)))  unsigned int  u32x4;

union BfPack { v16bf v; u32x4 u[2]; };

// ---------------------------------------------------------------------------
// Elementwise helpers
// ---------------------------------------------------------------------------

__global__ void wcvt_kernel(const float* __restrict__ src, __bf16* __restrict__ dst,
                            int cout, int cin, int k32, int n) {
    int t = blockIdx.x * blockDim.x + threadIdx.x;
    if (t >= n) return;
    int o = t / k32, c = t % k32;
    dst[t] = (c < cin) ? (__bf16)src[(size_t)o * cin + c] : (__bf16)0.f;
}

__global__ void gather_xyz_kernel(const float* __restrict__ src, const int* __restrict__ fps,
                                  float* __restrict__ dst, int B, int N, int M) {
    int t = blockIdx.x * blockDim.x + threadIdx.x;
    if (t >= B * M) return;
    int b = t / M;
    int s = fps[t];
    dst[(size_t)t * 3 + 0] = src[((size_t)b * N + s) * 3 + 0];
    dst[(size_t)t * 3 + 1] = src[((size_t)b * N + s) * 3 + 1];
    dst[(size_t)t * 3 + 2] = src[((size_t)b * N + s) * 3 + 2];
}

// Ball query: first NS ascending indices with d2 < r2 (== reference top_k(-scores)),
// padded with the first hit.
__global__ void ball_query_kernel(const float* __restrict__ xyz, const float* __restrict__ newxyz,
                                  int* __restrict__ idx, int B, int N, int M, int NS, float r2) {
    int t = blockIdx.x * blockDim.x + threadIdx.x;
    if (t >= B * M) return;
    int b = t / M;
    const float* q = newxyz + (size_t)t * 3;
    float qx = q[0], qy = q[1], qz = q[2];
    const float* p = xyz + (size_t)b * N * 3;
    int cnt = 0, first = 0;
    for (int n = 0; n < N && cnt < NS; ++n) {
        float dx = p[n * 3 + 0] - qx;
        float dy = p[n * 3 + 1] - qy;
        float dz = p[n * 3 + 2] - qz;
        float d2 = dx * dx + dy * dy + dz * dz;
        if (d2 < r2) {
            if (cnt == 0) first = n;
            idx[(size_t)t * NS + cnt] = n;
            ++cnt;
        }
    }
    for (; cnt < NS; ++cnt) idx[(size_t)t * NS + cnt] = first;
}

// ---------------------------------------------------------------------------
// Per-wave 16x16 GEMM layer via v_wmma_f32_16x16x32_bf16 + BN-affine + ReLU.
// act : LDS, ROWS(32) x K32 bf16 row-major (K32 % 32 == 0, zero-padded tail)
// outb: LDS, ROWS x Cout bf16 row-major
// W   : global, Cout x K32 bf16 row-major (pre-padded) -> B-tile columns are
//       contiguous weight-row segments (32B loads).
// 4 waves: wave&1 = row sub-tile, wave>>1 = col-tile parity. EXEC all-ones
// throughout (lane-dependent addresses only, wave-uniform control flow).
// ---------------------------------------------------------------------------
__device__ __forceinline__ void wave_gemm_bn_relu(const __bf16* __restrict__ act,
                                                  __bf16* __restrict__ outb,
                                                  const __bf16* __restrict__ W,
                                                  const float* __restrict__ gamma,
                                                  const float* __restrict__ beta,
                                                  int K32, int Cout, int wave, int lane) {
    const int rt   = wave & 1;
    const int cs   = wave >> 1;
    const int nct  = Cout >> 4;
    const int half = lane >> 4;    // lanes 16-31 hold K 16..31
    const int l15  = lane & 15;
    const __bf16* arow = act + (size_t)((rt << 4) + l15) * K32 + half * 16;
    for (int ct = cs; ct < nct; ct += 2) {
        const int o = (ct << 4) + l15;
        const __bf16* wrow = W + (size_t)o * K32 + half * 16;
        v8f acc = {0.f, 0.f, 0.f, 0.f, 0.f, 0.f, 0.f, 0.f};
        for (int k0 = 0; k0 < K32; k0 += 32) {
            BfPack A, Bm;
            A.u[0]  = *(const u32x4*)(arow + k0);        // A VGPR0-3: K kb..kb+7
            A.u[1]  = *(const u32x4*)(arow + k0 + 8);    // A VGPR4-7: K kb+8..kb+15
            Bm.u[0] = *(const u32x4*)(wrow + k0);        // B: 16 contiguous K of col o
            Bm.u[1] = *(const u32x4*)(wrow + k0 + 8);
            acc = __builtin_amdgcn_wmma_f32_16x16x32_bf16(
                false, A.v, false, Bm.v, (short)0, acc, false, false);
        }
        const float gm = gamma[o];
        const float bt = beta[o];
#pragma unroll
        for (int j = 0; j < 8; ++j) {
            const int m = (rt << 4) + j + half * 8;      // D VGPR j -> row j / j+8
            float v = acc[j] * gm + bt;
            v = v > 0.f ? v : 0.f;
            outb[(size_t)m * Cout + o] = (__bf16)v;
        }
    }
}

// ---------------------------------------------------------------------------
// Fused SA-MSG scale: group -> MLP(3 layers, WMMA) -> maxpool over NS.
// One workgroup handles TM query points (ROWS = TM*NS = 32 matrix rows).
// ---------------------------------------------------------------------------
template <int NS, int TM, int CIN, int FDIM, int C1, int C2, int C3>
__global__ __launch_bounds__(128) void sa_mlp_kernel(
    const float* __restrict__ xyz,     // (B, N, 3)
    const float* __restrict__ newxyz,  // (B, M, 3)
    const float* __restrict__ feat,    // (B, N, FDIM) or nullptr
    const int* __restrict__ idx,       // (B, M, NS)
    const __bf16* __restrict__ W1, const float* __restrict__ g1, const float* __restrict__ b1,
    const __bf16* __restrict__ W2, const float* __restrict__ g2, const float* __restrict__ b2,
    const __bf16* __restrict__ W3, const float* __restrict__ g3, const float* __restrict__ b3,
    float* __restrict__ outfeat,       // (B, M, out_stride), write cols [out_off, out_off+C3)
    int M, int N, int out_stride, int out_off) {
    constexpr int ROWS = 32;
    constexpr int K1 = (CIN + 31) & ~31;
    constexpr int K2 = (C1 + 31) & ~31;
    constexpr int K3 = (C2 + 31) & ~31;
    constexpr int KMAX = (K1 > K2 ? (K1 > K3 ? K1 : K3) : (K2 > K3 ? K2 : K3));

    extern __shared__ char smem[];
    __bf16* act  = (__bf16*)smem;                                 // ROWS x (<=KMAX)
    __bf16* outb = (__bf16*)(smem + (size_t)ROWS * KMAX * 2);     // ROWS x (<=maxCout)

    const int b   = blockIdx.y;
    const int m0  = blockIdx.x * TM;
    const int tid = threadIdx.x;
    const int wave = tid >> 5, lane = tid & 31;

    // Build grouped input (xyz-diff ++ feat), bf16, zero-padded to K1.
    for (int e = tid; e < ROWS * K1; e += blockDim.x) {
        int r = e / K1, c = e % K1;
        int mi = r / NS, n = r % NS;
        int m = m0 + mi;
        int src = idx[((size_t)b * M + m) * NS + n];
        float v = 0.f;
        if (c < 3)
            v = xyz[((size_t)b * N + src) * 3 + c] - newxyz[((size_t)b * M + m) * 3 + c];
        else if (c < 3 + FDIM)
            v = feat[((size_t)b * N + src) * FDIM + (c - 3)];
        act[e] = (__bf16)v;
    }
    __syncthreads();

    wave_gemm_bn_relu(act, outb, W1, g1, b1, K1, C1, wave, lane);
    __syncthreads();
    for (int e = tid; e < ROWS * K2; e += blockDim.x) {
        int r = e / K2, c = e % K2;
        act[e] = (c < C1) ? outb[(size_t)r * C1 + c] : (__bf16)0.f;
    }
    __syncthreads();

    wave_gemm_bn_relu(act, outb, W2, g2, b2, K2, C2, wave, lane);
    __syncthreads();
    for (int e = tid; e < ROWS * K3; e += blockDim.x) {
        int r = e / K3, c = e % K3;
        act[e] = (c < C2) ? outb[(size_t)r * C2 + c] : (__bf16)0.f;
    }
    __syncthreads();

    wave_gemm_bn_relu(act, outb, W3, g3, b3, K3, C3, wave, lane);
    __syncthreads();

    // Max-pool over NS samples per query point.
    for (int e = tid; e < TM * C3; e += blockDim.x) {
        int mi = e / C3, c = e % C3;
        float mx = -3.4e38f;
        for (int n = 0; n < NS; ++n) {
            float v = (float)outb[(size_t)(mi * NS + n) * C3 + c];
            mx = v > mx ? v : mx;
        }
        outfeat[((size_t)b * M + (m0 + mi)) * out_stride + out_off + c] = mx;
    }
}

// ---------------------------------------------------------------------------
// Plain WMMA GEMM for the final FC: out(R x O) = A(R x K, bf16) * W(O x K)^T
// One wave per 16x16 output tile; tile mapping is wave-uniform.
// ---------------------------------------------------------------------------
__global__ __launch_bounds__(128) void fc_gemm_kernel(const __bf16* __restrict__ A,
                                                      const __bf16* __restrict__ W,
                                                      float* __restrict__ out,
                                                      int R, int K, int O) {
    const int wave = threadIdx.x >> 5;
    const int lane = threadIdx.x & 31;
    const int tile = blockIdx.x * (blockDim.x >> 5) + wave;
    const int nct = O >> 4, nrt = R >> 4;
    const int rt = tile / nct, ct = tile % nct;
    if (rt >= nrt) return;
    const int half = lane >> 4, l15 = lane & 15;
    const __bf16* arow = A + (size_t)((rt << 4) + l15) * K + half * 16;
    const int o = (ct << 4) + l15;
    const __bf16* wrow = W + (size_t)o * K + half * 16;
    v8f acc = {0.f, 0.f, 0.f, 0.f, 0.f, 0.f, 0.f, 0.f};
    for (int k0 = 0; k0 < K; k0 += 32) {
        BfPack Aa, Bm;
        Aa.u[0] = *(const u32x4*)(arow + k0);
        Aa.u[1] = *(const u32x4*)(arow + k0 + 8);
        Bm.u[0] = *(const u32x4*)(wrow + k0);
        Bm.u[1] = *(const u32x4*)(wrow + k0 + 8);
        acc = __builtin_amdgcn_wmma_f32_16x16x32_bf16(
            false, Aa.v, false, Bm.v, (short)0, acc, false, false);
    }
#pragma unroll
    for (int j = 0; j < 8; ++j) {
        int m = (rt << 4) + j + half * 8;
        out[(size_t)m * O + o] = acc[j];
    }
}

// ---------------------------------------------------------------------------
// Head: bias add, sigmoid / roll-normalization, write final output.
// ---------------------------------------------------------------------------
__device__ __forceinline__ float sigmoidf_(float x) { return 1.f / (1.f + expf(-x)); }

__global__ void head_kernel(const float* __restrict__ raw, const float* __restrict__ bias,
                            float* __restrict__ out, int ngroups) {
    int g = blockIdx.x * blockDim.x + threadIdx.x;
    if (g >= ngroups) return;
    int row = g / 60;
    int og = (g % 60) * 8;
    const float* r = raw + (size_t)row * 480 + og;
    const float* bb = bias + og;
    float v[8];
#pragma unroll
    for (int i = 0; i < 8; ++i) v[i] = r[i] + bb[i];
    float* o = out + (size_t)row * 480 + og;
    o[0] = v[0];
    o[1] = sigmoidf_(v[1]);
    o[2] = sigmoidf_(v[2]);
    o[3] = sigmoidf_(v[3]);
    float nrm = sqrtf(v[4] * v[4] + v[5] * v[5]) + 1e-8f;
    o[4] = v[4] / nrm;
    o[5] = v[5] / nrm;
    o[6] = sigmoidf_(v[6]);
    o[7] = sigmoidf_(v[7]);
}

// ---------------------------------------------------------------------------
// Host orchestration (deterministic, capture-safe: kernels only).
// d_in order (jax tree-flatten of setup_inputs dict):
//   0 pointcloud, 1 idx0, 2 idx1, 3 idx2, 4 fc_b, 5 fc_w,
//   6.. per-layer triplets (W, beta, gamma) for sa[0][0][0..2], sa[0][1][..],
//   sa[1][0], sa[1][1], sa[2][0], sa[2][1].
// ---------------------------------------------------------------------------
extern "C" void kernel_launch(void* const* d_in, const int* in_sizes, int n_in,
                              void* d_out, int out_size, void* d_ws, size_t ws_size,
                              hipStream_t stream) {
    (void)in_sizes; (void)n_in; (void)out_size; (void)ws_size;
    const int B = 4;
    const float* pc   = (const float*)d_in[0];
    const int*   fps0 = (const int*)d_in[1];
    const int*   fps1 = (const int*)d_in[2];
    const int*   fps2 = (const int*)d_in[3];
    const float* fc_b = (const float*)d_in[4];
    const float* fc_w = (const float*)d_in[5];

    static const int LCO[19] = {16, 16, 32,  32, 32, 64,  64, 64, 128,  64, 96, 128,
                                128, 256, 512, 128, 256, 512, 480};
    static const int LCI[19] = {3, 16, 16,  3, 32, 32,  99, 64, 64,  99, 64, 96,
                                259, 128, 256, 259, 128, 256, 1024};

    char* ws = (char*)d_ws;
    size_t off = 0;
    auto alloc = [&](size_t bytes) -> size_t {
        size_t o = off;
        off = (off + bytes + 255) & ~(size_t)255;
        return o;
    };

    size_t wb[19];
    for (int i = 0; i < 19; ++i) {
        int k32 = (LCI[i] + 31) & ~31;
        wb[i] = alloc((size_t)LCO[i] * k32 * sizeof(__bf16));
    }
    size_t nx0  = alloc((size_t)B * 2048 * 3 * 4);
    size_t nx1  = alloc((size_t)B * 512 * 3 * 4);
    size_t nx2  = alloc((size_t)B * 128 * 3 * 4);
    size_t bq00 = alloc((size_t)B * 2048 * 16 * 4);
    size_t bq01 = alloc((size_t)B * 2048 * 32 * 4);
    size_t bq10 = alloc((size_t)B * 512 * 16 * 4);
    size_t bq11 = alloc((size_t)B * 512 * 32 * 4);
    size_t bq20 = alloc((size_t)B * 128 * 16 * 4);
    size_t bq21 = alloc((size_t)B * 128 * 32 * 4);
    size_t f0   = alloc((size_t)B * 2048 * 96 * 4);
    size_t f1   = alloc((size_t)B * 512 * 256 * 4);
    size_t f2   = alloc((size_t)B * 128 * 1024 * 4);
    size_t f2b  = alloc((size_t)B * 128 * 1024 * sizeof(__bf16));
    size_t raw  = alloc((size_t)B * 128 * 480 * 4);

    auto Wp = [&](int i) { return (const __bf16*)(ws + wb[i]); };
    auto Bp = [&](int i) { return (const float*)d_in[6 + 3 * i + 1]; };  // beta
    auto Gp = [&](int i) { return (const float*)d_in[6 + 3 * i + 2]; };  // gamma

    // Convert + K-pad all weights to bf16 (deterministic, every call).
    for (int i = 0; i < 19; ++i) {
        int k32 = (LCI[i] + 31) & ~31;
        const float* src = (i == 18) ? fc_w : (const float*)d_in[6 + 3 * i];
        int n = LCO[i] * k32;
        wcvt_kernel<<<dim3((n + 255) / 256), 256, 0, stream>>>(
            src, (__bf16*)(ws + wb[i]), LCO[i], LCI[i], k32, n);
    }

    float* pnx0 = (float*)(ws + nx0);
    float* pnx1 = (float*)(ws + nx1);
    float* pnx2 = (float*)(ws + nx2);

    // ---- Level 0: N=8192 -> M=2048, CIN=3 -------------------------------
    gather_xyz_kernel<<<dim3((B * 2048 + 255) / 256), 256, 0, stream>>>(pc, fps0, pnx0, B, 8192, 2048);
    ball_query_kernel<<<dim3((B * 2048 + 127) / 128), 128, 0, stream>>>(
        pc, pnx0, (int*)(ws + bq00), B, 8192, 2048, 16, 0.01f * 0.01f);
    ball_query_kernel<<<dim3((B * 2048 + 127) / 128), 128, 0, stream>>>(
        pc, pnx0, (int*)(ws + bq01), B, 8192, 2048, 32, 0.03f * 0.03f);
    sa_mlp_kernel<16, 2, 3, 0, 16, 16, 32><<<dim3(1024, B), 128, 4096, stream>>>(
        pc, pnx0, nullptr, (int*)(ws + bq00),
        Wp(0), Gp(0), Bp(0), Wp(1), Gp(1), Bp(1), Wp(2), Gp(2), Bp(2),
        (float*)(ws + f0), 2048, 8192, 96, 0);
    sa_mlp_kernel<32, 1, 3, 0, 32, 32, 64><<<dim3(2048, B), 128, 6144, stream>>>(
        pc, pnx0, nullptr, (int*)(ws + bq01),
        Wp(3), Gp(3), Bp(3), Wp(4), Gp(4), Bp(4), Wp(5), Gp(5), Bp(5),
        (float*)(ws + f0), 2048, 8192, 96, 32);

    // ---- Level 1: N=2048 -> M=512, CIN=99 -------------------------------
    gather_xyz_kernel<<<dim3((B * 512 + 255) / 256), 256, 0, stream>>>(pnx0, fps1, pnx1, B, 2048, 512);
    ball_query_kernel<<<dim3((B * 512 + 127) / 128), 128, 0, stream>>>(
        pnx0, pnx1, (int*)(ws + bq10), B, 2048, 512, 16, 0.025f * 0.025f);
    ball_query_kernel<<<dim3((B * 512 + 127) / 128), 128, 0, stream>>>(
        pnx0, pnx1, (int*)(ws + bq11), B, 2048, 512, 32, 0.05f * 0.05f);
    sa_mlp_kernel<16, 2, 99, 96, 64, 64, 128><<<dim3(256, B), 128, 16384, stream>>>(
        pnx0, pnx1, (const float*)(ws + f0), (int*)(ws + bq10),
        Wp(6), Gp(6), Bp(6), Wp(7), Gp(7), Bp(7), Wp(8), Gp(8), Bp(8),
        (float*)(ws + f1), 512, 2048, 256, 0);
    sa_mlp_kernel<32, 1, 99, 96, 64, 96, 128><<<dim3(512, B), 128, 16384, stream>>>(
        pnx0, pnx1, (const float*)(ws + f0), (int*)(ws + bq11),
        Wp(9), Gp(9), Bp(9), Wp(10), Gp(10), Bp(10), Wp(11), Gp(11), Bp(11),
        (float*)(ws + f1), 512, 2048, 256, 128);

    // ---- Level 2: N=512 -> M=128, CIN=259 -------------------------------
    gather_xyz_kernel<<<dim3((B * 128 + 255) / 256), 256, 0, stream>>>(pnx1, fps2, pnx2, B, 512, 128);
    ball_query_kernel<<<dim3((B * 128 + 127) / 128), 128, 0, stream>>>(
        pnx1, pnx2, (int*)(ws + bq20), B, 512, 128, 16, 0.1f * 0.1f);
    ball_query_kernel<<<dim3((B * 128 + 127) / 128), 128, 0, stream>>>(
        pnx1, pnx2, (int*)(ws + bq21), B, 512, 128, 32, 0.15f * 0.15f);
    sa_mlp_kernel<16, 2, 259, 256, 128, 256, 512><<<dim3(64, B), 128, 51200, stream>>>(
        pnx1, pnx2, (const float*)(ws + f1), (int*)(ws + bq20),
        Wp(12), Gp(12), Bp(12), Wp(13), Gp(13), Bp(13), Wp(14), Gp(14), Bp(14),
        (float*)(ws + f2), 128, 512, 1024, 0);
    sa_mlp_kernel<32, 1, 259, 256, 128, 256, 512><<<dim3(128, B), 128, 51200, stream>>>(
        pnx1, pnx2, (const float*)(ws + f1), (int*)(ws + bq21),
        Wp(15), Gp(15), Bp(15), Wp(16), Gp(16), Bp(16), Wp(17), Gp(17), Bp(17),
        (float*)(ws + f2), 128, 512, 1024, 512);

    // ---- FC (512x480, K=1024) + head ------------------------------------
    {
        int n = B * 128 * 1024;
        wcvt_kernel<<<dim3((n + 255) / 256), 256, 0, stream>>>(
            (const float*)(ws + f2), (__bf16*)(ws + f2b), B * 128, 1024, 1024, n);
    }
    fc_gemm_kernel<<<dim3(240), 128, 0, stream>>>(
        (const __bf16*)(ws + f2b), Wp(18), (float*)(ws + raw), 512, 1024, 480);
    {
        int ng = B * 128 * 60;
        head_kernel<<<dim3((ng + 255) / 256), 256, 0, stream>>>(
            (const float*)(ws + raw), fc_b, (float*)d_out, ng);
    }
}
